// Block_65618510348705
// MI455X (gfx1250) — matrix operand order repaired and checked by reading, not compiled, not run on hardware
//
#include <hip/hip_runtime.h>
#include <cmath>

// ---------------------------------------------------------------------------
// Types for CDNA5 WMMA (wave32): bf16 A/B (16 halves = 8 VGPRs), f32 C/D (8 VGPRs)
// ---------------------------------------------------------------------------
typedef __attribute__((ext_vector_type(16))) __bf16 v16bf;
typedef __attribute__((ext_vector_type(8)))  float  v8f;
typedef unsigned short u16;

union ABPack { uint4 q[2]; v16bf v; };

static __device__ inline v8f wmma_bf16(v16bf a, v16bf b, v8f c) {
    // 8 args: (neg_a, A, neg_b, B, c_mod, C, reuse_a, reuse_b)
    return __builtin_amdgcn_wmma_f32_16x16x32_bf16(false, a, false, b, (short)0, c, false, false);
}

static __device__ inline v8f v8f_zero() {
    v8f z = {0.f, 0.f, 0.f, 0.f, 0.f, 0.f, 0.f, 0.f};
    return z;
}

// round-to-nearest-even f32 -> bf16 bits
static __device__ inline u16 f2bf(float f) {
    unsigned int u = __builtin_bit_cast(unsigned int, f);
    u += 0x7FFFu + ((u >> 16) & 1u);
    return (u16)(u >> 16);
}

// ---------------------------------------------------------------------------
// Problem constants
// ---------------------------------------------------------------------------
#define BB   8
#define NN   2048
#define CC   768
#define HH   3072
#define MTOT (BB * NN)              // 16384 rows
#define SC2  (0.125f * 1.44269504088896f)   // (1/sqrt(64)) * log2(e)

// workspace byte offsets (all 256-aligned)
#define OFF_YB   ((size_t)0)              // bf16 [B,N,C]      25,165,824 B
#define OFF_YTB  ((size_t)25165824)       // bf16 [B,C,N]      25,165,824 B
#define OFF_X1   ((size_t)50331648)       // f32  [B,N,C]      50,331,648 B
#define OFF_ZB   ((size_t)100663296)      // bf16 [B,N,C]      25,165,824 B
#define OFF_HB   ((size_t)125829120)      // bf16 [B*N,H]     100,663,296 B
#define OFF_W1B  ((size_t)226492416)      // bf16 [H,C]         4,718,592 B
#define OFF_W2B  ((size_t)231211008)      // bf16 [C,H]         4,718,592 B

// ---------------------------------------------------------------------------
// f32 -> bf16 weight conversion
// ---------------------------------------------------------------------------
__global__ __launch_bounds__(256) void cvt_kernel(const float* __restrict__ src,
                                                  u16* __restrict__ dst, int n) {
    for (int i = blockIdx.x * 256 + threadIdx.x; i < n; i += gridDim.x * 256)
        dst[i] = f2bf(src[i]);
}

// ---------------------------------------------------------------------------
// LayerNorm: one wave32 per 768-wide row; optional transposed bf16 output
// ---------------------------------------------------------------------------
__global__ __launch_bounds__(256) void ln_kernel(const float* __restrict__ x,
                                                 const float* __restrict__ w,
                                                 const float* __restrict__ bvec,
                                                 u16* __restrict__ out,
                                                 u16* __restrict__ outT) {
    const int wv   = threadIdx.x >> 5;
    const int lane = threadIdx.x & 31;
    const int row  = blockIdx.x * 8 + wv;
    const float* xr = x + (size_t)row * CC;

    float vals[24];
    float s = 0.f, ss = 0.f;
    #pragma unroll
    for (int i = 0; i < 24; i++) {
        float v = xr[lane + i * 32];
        vals[i] = v; s += v; ss += v * v;
    }
    #pragma unroll
    for (int m = 1; m < 32; m <<= 1) {
        s  += __shfl_xor(s,  m, 32);
        ss += __shfl_xor(ss, m, 32);
    }
    const float mu = s * (1.f / CC);
    const float ri = rsqrtf(ss * (1.f / CC) - mu * mu + 1e-5f);
    const int b = row >> 11, n = row & (NN - 1);
    #pragma unroll
    for (int i = 0; i < 24; i++) {
        int c = lane + i * 32;
        u16 bits = f2bf((vals[i] - mu) * ri * w[c] + bvec[c]);
        out[(size_t)row * CC + c] = bits;
        if (outT) outT[((size_t)(b * CC + c)) * NN + n] = bits;
    }
}

// ---------------------------------------------------------------------------
// Attention: workgroup (8 waves) per 16-query tile. Single QK^T pass: the full
// 16x2048 score strip (scaled log2 domain) lives in LDS (128 KB of the 320 KB
// CDNA5 WGP pool). Softmax stats are a separate shuffle-free phase over the
// strip (each thread owns 1 row x 8 16-col chunks, 256 apart), and the strip is
// overwritten in place with P = exp2(S - M), so pass 2 only converts to bf16.
//   q = k = v = y = LN1(x);  x1 = x + softmax(q kT * scale) v
// ---------------------------------------------------------------------------
__global__ __launch_bounds__(256) void attn_kernel(const u16* __restrict__ yb,
                                                   const u16* __restrict__ ytb,
                                                   const float* __restrict__ x,
                                                   float* __restrict__ x1) {
    __shared__ __align__(16) u16   QPs[16 * CC];    // 24576 B: Q tile (pass1) / P bf16 (pass2)
    __shared__ __align__(16) float Sf[16 * NN];     // 131072 B: scores, then P (f32)
    __shared__ float partM[16][16], partL[16][16];
    __shared__ float combL[16];

    const int wv = threadIdx.x >> 5, lane = threadIdx.x & 31;
    const int g  = lane >> 4,        ln   = lane & 15;
    const int b  = blockIdx.x >> 7;
    const int m0 = (blockIdx.x & 127) * 16;

    // stage 16x768 bf16 Q tile into LDS (contiguous copy)
    {
        const uint4* src = (const uint4*)(yb + ((size_t)(b * NN + m0)) * CC);
        uint4* dst = (uint4*)QPs;
        for (int t = threadIdx.x; t < (16 * CC) / 8; t += 256) dst[t] = src[t];
    }
    __syncthreads();

    const u16* Qrow = QPs + ln * CC;

    // ---- pass 1: pure WMMA stream, S = Q K^T * scale -> LDS strip ----
    for (int j = wv; j < 128; j += 8) {
        v8f acc = v8f_zero();
        const u16* Krow = yb + ((size_t)(b * NN + j * 16 + ln)) * CC;
        #pragma unroll 4
        for (int kc = 0; kc < CC; kc += 32) {
            ABPack a, bm;
            a.q[0]  = *(const uint4*)(Qrow + kc + g * 8);
            a.q[1]  = *(const uint4*)(Qrow + kc + g * 8 + 16);
            bm.q[0] = *(const uint4*)(Krow + kc + g * 8);
            bm.q[1] = *(const uint4*)(Krow + kc + g * 8 + 16);
            acc = wmma_bf16(a.v, bm.v, acc);
        }
        #pragma unroll
        for (int i = 0; i < 8; i++)
            Sf[(g * 8 + i) * NN + j * 16 + ln] = acc[i] * SC2;   // scaled, log2 domain
    }
    __syncthreads();

    // ---- softmax stats phase (no cross-lane shuffles) ----
    const int prow = threadIdx.x >> 4;              // row 0..15
    const int pch  = threadIdx.x & 15;              // 16-col chunk id
    const int pcol = pch * 16;
    float* Srow = Sf + prow * NN;

    float mloc = -3.0e38f;
    for (int kb = 0; kb < 8; kb++)
        #pragma unroll
        for (int e = 0; e < 16; e++)
            mloc = fmaxf(mloc, Srow[kb * 256 + pcol + e]);
    partM[prow][pch] = mloc;
    __syncthreads();

    float M = partM[prow][0];
    #pragma unroll
    for (int k = 1; k < 16; k++) M = fmaxf(M, partM[prow][k]);

    float lloc = 0.f;
    for (int kb = 0; kb < 8; kb++)
        #pragma unroll
        for (int e = 0; e < 16; e++) {
            float p = exp2f(Srow[kb * 256 + pcol + e] - M);
            Srow[kb * 256 + pcol + e] = p;          // strip now holds P (f32)
            lloc += p;
        }
    partL[prow][pch] = lloc;
    __syncthreads();
    if (pch == 0) {
        float L = 0.f;
        #pragma unroll
        for (int k = 0; k < 16; k++) L += partL[prow][k];
        combL[prow] = L;
    }
    __syncthreads();        // Q tile dead; QPs becomes bf16 P staging

    // ---- pass 2: O += P * V over 256-key superblocks ----
    u16* Ps = QPs;                                  // 16 x 256 bf16 staging
    v8f o[6];
    #pragma unroll
    for (int cc = 0; cc < 6; cc++) o[cc] = v8f_zero();

    for (int kb = 0; kb < 8; kb++) {
        #pragma unroll
        for (int e = 0; e < 16; e++)                // convert-only P staging
            Ps[prow * 256 + pcol + e] = f2bf(Srow[kb * 256 + pcol + e]);
        __syncthreads();

        const u16* Prow = Ps + ln * 256;
        #pragma unroll 2
        for (int kk = 0; kk < 8; kk++) {
            ABPack a;
            a.q[0] = *(const uint4*)(Prow + kk * 32 + g * 8);
            a.q[1] = *(const uint4*)(Prow + kk * 32 + g * 8 + 16);
            #pragma unroll
            for (int cc = 0; cc < 6; cc++) {
                const int n = wv * 96 + cc * 16 + ln;
                const u16* bp = ytb + ((size_t)(b * CC + n)) * NN + kb * 256 + kk * 32 + g * 8;
                ABPack bm;
                bm.q[0] = *(const uint4*)bp;
                bm.q[1] = *(const uint4*)(bp + 16);
                o[cc] = wmma_bf16(a.v, bm.v, o[cc]);
            }
        }
        __syncthreads();
    }

    // ---- normalize + residual ----
    float linv[8];
    #pragma unroll
    for (int i = 0; i < 8; i++) linv[i] = 1.0f / combL[g * 8 + i];
    #pragma unroll
    for (int cc = 0; cc < 6; cc++) {
        #pragma unroll
        for (int i = 0; i < 8; i++) {
            const int row = m0 + g * 8 + i;
            const size_t idx = ((size_t)(b * NN + row)) * CC + wv * 96 + cc * 16 + ln;
            x1[idx] = x[idx] + o[cc][i] * linv[i];
        }
    }
}

// ---------------------------------------------------------------------------
// MLP GEMM 1: h = gelu(z @ W1^T + b1), z bf16 [M,768], W1 bf16 [3072,768]
// workgroup tile 64(M) x 256(N); wave tile 32 x 64 (B fragments reused 2x)
// ---------------------------------------------------------------------------
__global__ __launch_bounds__(256) void mlp1_kernel(const u16* __restrict__ zb,
                                                   const u16* __restrict__ w1,
                                                   const float* __restrict__ bias,
                                                   u16* __restrict__ hb) {
    const int wv = threadIdx.x >> 5, lane = threadIdx.x & 31;
    const int g  = lane >> 4,        ln   = lane & 15;
    const int bm = blockIdx.x & 255, bn = blockIdx.x >> 8;   // bn in 0..11
    const int r0 = bm * 64  + (wv & 1) * 32;
    const int c0 = bn * 256 + (wv >> 1) * 64;

    v8f acc[2][4];
    #pragma unroll
    for (int s = 0; s < 2; s++)
        #pragma unroll
        for (int cc = 0; cc < 4; cc++) acc[s][cc] = v8f_zero();

    const u16* arow0 = zb + (size_t)(r0 + ln) * CC;
    const u16* arow1 = zb + (size_t)(r0 + 16 + ln) * CC;
    #pragma unroll 2
    for (int kc = 0; kc < CC; kc += 32) {
        ABPack a[2];
        a[0].q[0] = *(const uint4*)(arow0 + kc + g * 8);
        a[0].q[1] = *(const uint4*)(arow0 + kc + g * 8 + 16);
        a[1].q[0] = *(const uint4*)(arow1 + kc + g * 8);
        a[1].q[1] = *(const uint4*)(arow1 + kc + g * 8 + 16);
        #pragma unroll
        for (int cc = 0; cc < 4; cc++) {
            const u16* bp = w1 + (size_t)(c0 + cc * 16 + ln) * CC + kc + g * 8;
            ABPack bm2;
            bm2.q[0] = *(const uint4*)bp;
            bm2.q[1] = *(const uint4*)(bp + 16);
            acc[0][cc] = wmma_bf16(a[0].v, bm2.v, acc[0][cc]);
            acc[1][cc] = wmma_bf16(a[1].v, bm2.v, acc[1][cc]);
        }
    }
    #pragma unroll
    for (int s = 0; s < 2; s++) {
        #pragma unroll
        for (int cc = 0; cc < 4; cc++) {
            const int col = c0 + cc * 16 + ln;
            const float bv = bias[col];
            #pragma unroll
            for (int i = 0; i < 8; i++) {
                const int row = r0 + s * 16 + g * 8 + i;
                float v  = acc[s][cc][i] + bv;
                float ge = 0.5f * v * (1.f + erff(v * 0.70710678118f));   // exact gelu
                hb[(size_t)row * HH + col] = f2bf(ge);
            }
        }
    }
}

// ---------------------------------------------------------------------------
// MLP GEMM 2: out = x1 + h @ W2^T + b2, h bf16 [M,3072], W2 bf16 [768,3072]
// ---------------------------------------------------------------------------
__global__ __launch_bounds__(256) void mlp2_kernel(const u16* __restrict__ hb,
                                                   const u16* __restrict__ w2,
                                                   const float* __restrict__ bias,
                                                   const float* __restrict__ x1,
                                                   float* __restrict__ out) {
    const int wv = threadIdx.x >> 5, lane = threadIdx.x & 31;
    const int g  = lane >> 4,        ln   = lane & 15;
    const int bm = blockIdx.x & 255, bn = blockIdx.x >> 8;   // bn in 0..2
    const int r0 = bm * 64  + (wv & 1) * 32;
    const int c0 = bn * 256 + (wv >> 1) * 64;

    v8f acc[2][4];
    #pragma unroll
    for (int s = 0; s < 2; s++)
        #pragma unroll
        for (int cc = 0; cc < 4; cc++) acc[s][cc] = v8f_zero();

    const u16* arow0 = hb + (size_t)(r0 + ln) * HH;
    const u16* arow1 = hb + (size_t)(r0 + 16 + ln) * HH;
    #pragma unroll 2
    for (int kc = 0; kc < HH; kc += 32) {
        ABPack a[2];
        a[0].q[0] = *(const uint4*)(arow0 + kc + g * 8);
        a[0].q[1] = *(const uint4*)(arow0 + kc + g * 8 + 16);
        a[1].q[0] = *(const uint4*)(arow1 + kc + g * 8);
        a[1].q[1] = *(const uint4*)(arow1 + kc + g * 8 + 16);
        #pragma unroll
        for (int cc = 0; cc < 4; cc++) {
            const u16* bp = w2 + (size_t)(c0 + cc * 16 + ln) * HH + kc + g * 8;
            ABPack bm2;
            bm2.q[0] = *(const uint4*)bp;
            bm2.q[1] = *(const uint4*)(bp + 16);
            acc[0][cc] = wmma_bf16(a[0].v, bm2.v, acc[0][cc]);
            acc[1][cc] = wmma_bf16(a[1].v, bm2.v, acc[1][cc]);
        }
    }
    #pragma unroll
    for (int s = 0; s < 2; s++) {
        #pragma unroll
        for (int cc = 0; cc < 4; cc++) {
            const int col = c0 + cc * 16 + ln;
            const float bv = bias[col];
            #pragma unroll
            for (int i = 0; i < 8; i++) {
                const int row = r0 + s * 16 + g * 8 + i;
                const size_t idx = (size_t)row * CC + col;
                out[idx] = x1[idx] + acc[s][cc][i] + bv;
            }
        }
    }
}

// ---------------------------------------------------------------------------
// launch
// ---------------------------------------------------------------------------
extern "C" void kernel_launch(void* const* d_in, const int* in_sizes, int n_in,
                              void* d_out, int out_size, void* d_ws, size_t ws_size,
                              hipStream_t stream) {
    const float* x     = (const float*)d_in[0];
    const float* ln1_w = (const float*)d_in[1];
    const float* ln1_b = (const float*)d_in[2];
    const float* ln2_w = (const float*)d_in[3];
    const float* ln2_b = (const float*)d_in[4];
    const float* fc1_w = (const float*)d_in[5];
    const float* fc1_b = (const float*)d_in[6];
    const float* fc2_w = (const float*)d_in[7];
    const float* fc2_b = (const float*)d_in[8];

    char* ws = (char*)d_ws;
    u16*   yb  = (u16*)  (ws + OFF_YB);
    u16*   ytb = (u16*)  (ws + OFF_YTB);
    float* x1  = (float*)(ws + OFF_X1);
    u16*   zb  = (u16*)  (ws + OFF_ZB);
    u16*   hb  = (u16*)  (ws + OFF_HB);
    u16*   w1b = (u16*)  (ws + OFF_W1B);
    u16*   w2b = (u16*)  (ws + OFF_W2B);

    const int wn = HH * CC;   // 2,359,296 elements per weight
    cvt_kernel<<<1024, 256, 0, stream>>>(fc1_w, w1b, wn);
    cvt_kernel<<<1024, 256, 0, stream>>>(fc2_w, w2b, wn);

    ln_kernel<<<MTOT / 8, 256, 0, stream>>>(x, ln1_w, ln1_b, yb, ytb);

    attn_kernel<<<BB * (NN / 16), 256, 0, stream>>>(yb, ytb, x, x1);

    ln_kernel<<<MTOT / 8, 256, 0, stream>>>(x1, ln2_w, ln2_b, zb, (u16*)nullptr);

    mlp1_kernel<<<(MTOT / 64) * (HH / 256), 256, 0, stream>>>(zb, w1b, fc1_b, hb);

    mlp2_kernel<<<(MTOT / 64) * (CC / 256), 256, 0, stream>>>(hb, w2b, fc2_b, x1,
                                                              (float*)d_out);
}
